// RGCN_26585847562311
// MI455X (gfx1250) — compile-verified, matrix operand order checked by reading, not run on hardware
//
#include <hip/hip_runtime.h>

// ---------------- types for WMMA (CDNA5 / gfx1250, wave32) ----------------
typedef __attribute__((ext_vector_type(16))) __bf16 v16bf;
typedef __attribute__((ext_vector_type(8)))  __bf16 v8bf;
typedef __attribute__((ext_vector_type(8)))  float  v8f;

#define T_  5

// Load one 16x32 bf16 fragment (A-layout; B uses same layout on transposed
// weights).  Lane L (0..15) holds row L, K-chunks {0..7, 16..23};
// lane L+16 holds row L, K-chunks {8..15, 24..31}.  Two 16-byte loads.
__device__ __forceinline__ v16bf load_frag(const __bf16* tile, int stride, int lane) {
    int row  = lane & 15;
    int half = (lane >> 4) << 3;           // 0 or 8 elements
    const __bf16* p = tile + row * stride + half;
    v8bf c0 = *reinterpret_cast<const v8bf*>(p);
    v8bf c1 = *reinterpret_cast<const v8bf*>(p + 16);
    v16bf r;
#pragma unroll
    for (int x = 0; x < 8; ++x) { r[x] = c0[x]; r[8 + x] = c1[x]; }
    return r;
}

// ---- CDNA5 async global->LDS copy (ASYNCcnt-tracked, bypasses VGPRs) ----
__device__ __forceinline__ void async_g2l_b128(void* lds_dst, const float* gsrc) {
    unsigned loff = (unsigned)(unsigned long long)
        (__attribute__((address_space(3))) void*)lds_dst;
    asm volatile("global_load_async_to_lds_b128 %0, %1, off"
                 :: "v"(loff), "v"(gsrc) : "memory");
}
__device__ __forceinline__ void wait_asynccnt0() {
    asm volatile("s_wait_asynccnt 0" ::: "memory");
}

// ---------------- setup kernels (tiny) ----------------

// Wt[n*K + k] = (bf16) W[k*N + n]
__global__ void k_transpose_bf16(const float* __restrict__ W, __bf16* __restrict__ Wt,
                                 int K, int N) {
    int idx = blockIdx.x * blockDim.x + threadIdx.x;
    if (idx < K * N) {
        int k = idx / N, n = idx % N;
        Wt[n * K + k] = (__bf16)W[idx];
    }
}

// Wcat[n*512 + k] = W1[k*256 + n]   (k < 256 half: max_pool path)
__global__ void k_wcat_a(const float* __restrict__ W1, __bf16* __restrict__ Wcat) {
    int idx = blockIdx.x * 256 + threadIdx.x;      // 65536
    int n = idx >> 8, k = idx & 255;
    Wcat[n * 512 + k] = (__bf16)W1[k * 256 + n];
}

// Wc = W_img @ W1[256:512]  folded directly (transposed, bf16) into Wcat upper half
__global__ void k_wcat_b(const float* __restrict__ Wimg, const float* __restrict__ W1,
                         __bf16* __restrict__ Wcat) {
    __shared__ float wrow[256];
    int k = blockIdx.x, n = threadIdx.x;
    wrow[n] = Wimg[k * 256 + n];
    __syncthreads();
    float s = 0.f;
    for (int m = 0; m < 256; ++m) s = fmaf(wrow[m], W1[(256 + m) * 256 + n], s);
    Wcat[n * 512 + 256 + k] = (__bf16)s;
}

// fi2p / fsslp = x @ W_img + b_img     (16 x 256 each)
__global__ void k_small_proj(const float* __restrict__ fi2, const float* __restrict__ fssl,
                             const float* __restrict__ Wimg, const float* __restrict__ bimg,
                             float* __restrict__ fi2p, float* __restrict__ fsslp) {
    __shared__ float xr[256];
    int b = blockIdx.x, sel = blockIdx.y, n = threadIdx.x;
    const float* x = sel ? fssl : fi2;
    xr[n] = x[b * 256 + n];
    __syncthreads();
    float s = bimg[n];
    for (int m = 0; m < 256; ++m) s = fmaf(xr[m], Wimg[m * 256 + n], s);
    (sel ? fsslp : fi2p)[b * 256 + n] = s;
}

// h_base[b][n] = b1[n] + fi2p@W1[512:768] + fsslp@W1[768:1024] + b_img@W1[256:512]
__global__ void k_hbase(const float* __restrict__ fi2p, const float* __restrict__ fsslp,
                        const float* __restrict__ bimg, const float* __restrict__ W1,
                        const float* __restrict__ b1, float* __restrict__ hbase) {
    __shared__ float a2[256], a3[256], ab[256];
    int b = blockIdx.x, n = threadIdx.x;
    a2[n] = fi2p[b * 256 + n];
    a3[n] = fsslp[b * 256 + n];
    ab[n] = bimg[n];
    __syncthreads();
    float s = b1[n];
    for (int m = 0; m < 256; ++m) {
        s = fmaf(a2[m], W1[(512 + m) * 256 + n], s);
        s = fmaf(a3[m], W1[(768 + m) * 256 + n], s);
        s = fmaf(ab[m], W1[(256 + m) * 256 + n], s);
    }
    hbase[b * 256 + n] = s;
}

// ---------------- WMMA GEMM: Y[r,n] = X[r,:] @ W[:,n] + bias[n] ----------------
// X: [1024,256] f32, Wt: [256n][256k] bf16 (pre-transposed), 16 rows / block.
// Each wave owns a 16x32 output strip: A fragment loaded once per k-block,
// two WMMAs against two B fragments (A reuse).
__global__ void k_gemm_act(const float* __restrict__ X, const __bf16* __restrict__ Wt,
                           const float* __restrict__ bias, float* __restrict__ Y) {
    __shared__ __align__(16) __bf16 Abuf[16 * 256];
    int tid = threadIdx.x;
    int r0  = blockIdx.x * 16;
    for (int idx = tid; idx < 16 * 256; idx += 256)
        Abuf[idx] = (__bf16)X[r0 * 256 + idx];
    __syncthreads();

    int wave = tid >> 5, lane = tid & 31;
    int n0 = wave * 32;
    v8f acc0 = {}, acc1 = {};
#pragma unroll
    for (int kb = 0; kb < 8; ++kb) {
        int k0 = kb * 32;
        v16bf a  = load_frag(Abuf + k0, 256, lane);
        v16bf b0 = load_frag(Wt + n0 * 256 + k0, 256, lane);
        v16bf b1 = load_frag(Wt + (n0 + 16) * 256 + k0, 256, lane);
        acc0 = __builtin_amdgcn_wmma_f32_16x16x32_bf16(
            false, a, false, b0, (short)0, acc0, false, false);
        acc1 = __builtin_amdgcn_wmma_f32_16x16x32_bf16(
            false, a, false, b1, (short)0, acc1, false, false);
    }
    int rbase = (lane >> 4) << 3;
#pragma unroll
    for (int sub = 0; sub < 2; ++sub) {
        int col  = n0 + sub * 16 + (lane & 15);
        float bv = bias[col];
        const v8f& acc = sub ? acc1 : acc0;
#pragma unroll
        for (int rr = 0; rr < 8; ++rr)
            Y[(r0 + rbase + rr) * 256 + col] = acc[rr] + bv;
    }
}

// ---------------- per-step aggregation: prop += relu(msg + agg) ----------------
__global__ void k_agg(const float* __restrict__ msg, const int* __restrict__ mask,
                      float* __restrict__ prop) {
    __shared__ float smask[64];
    int b = blockIdx.x >> 6, i = blockIdx.x & 63;
    int h = threadIdx.x;
    if (h < 64) smask[h] = (float)mask[(b * 64 + i) * 64 + h];
    __syncthreads();
    const float* mb = msg + b * 64 * 256;
    float mi = mb[i * 256 + h];
    float acc = 0.f;
    for (int j = 0; j < 64; ++j) {
        float mj = mb[j * 256 + h];
        float fe = fmaxf(mi + mj, 0.f);
        acc = fmaf(fe * mj, smask[j], acc);
    }
    int idx = (b * 64 + i) * 256 + h;
    prop[idx] += fmaxf(mi + acc, 0.f);
}

// ---------------- fused edge kernel ----------------
// One block = 16 rows (fixed b,i; j0..j0+15).  A = [max_pool(256) | fimg1(256)],
// hidden = relu(A @ Wcat + h_base), out = (hidden @ W2 + b2) * mask.
// feat_img_1 tile is moved global->LDS with CDNA5 async-to-LDS copies that
// overlap with the max_pool regeneration (msgs are L2-hot).
__global__ void k_edge(const float* __restrict__ msgs,   // [5][16][64][256]
                       const float* __restrict__ fimg1,  // [16][64][64][256]
                       const __bf16* __restrict__ Wcat,  // [256n][512k]
                       const float* __restrict__ hbase,  // [16][256]
                       const float* __restrict__ W2,     // [256][8]
                       const float* __restrict__ b2,     // [8]
                       const int*   __restrict__ mask,   // [16][64][64]
                       float* __restrict__ out) {        // [16][64][64][8]
    __shared__ __align__(16) __bf16 Abuf[16 * 512];
    __shared__ __align__(16) float  stagef[16 * 256];
    __shared__ float hid[16 * 256];
    __shared__ float w2s[256 * 8];

    int tid = threadIdx.x;
    int blk = blockIdx.x;          // 4096 blocks
    int b   = blk >> 8;
    int rem = blk & 255;
    int i   = rem >> 2;
    int j0  = (rem & 3) << 4;

    // --- issue async global->LDS copies for the 16 contiguous fimg1 rows ---
    const float* gbase = fimg1 + ((size_t)((b * 64 + i) * 64 + j0)) * 256;
#pragma unroll
    for (int c = tid; c < 1024; c += 256)          // 1024 x 16B chunks
        async_g2l_b128(stagef + c * 4, gbase + c * 4);

    for (int idx = tid; idx < 2048; idx += 256) w2s[idx] = W2[idx];

    // --- overlap: regenerate max_pool half of the A tile from msgs ---
    for (int idx = tid; idx < 4096; idx += 256) {
        int row = idx >> 8, h = idx & 255;
        int j = j0 + row;
        float mx = -3.0e38f;
#pragma unroll
        for (int t = 0; t < T_; ++t) {
            float a = msgs[(((t * 16 + b) * 64 + i) * 256) + h]
                    + msgs[(((t * 16 + b) * 64 + j) * 256) + h];
            mx = fmaxf(mx, a);
        }
        Abuf[row * 512 + h] = (__bf16)fmaxf(mx, 0.f);  // max_t relu == relu(max_t)
    }

    wait_asynccnt0();              // this wave's async copies landed in LDS
    __syncthreads();               // all waves' copies + max_pool done

    // --- convert staged fp32 tile to bf16 fragment layout ---
    for (int idx = tid; idx < 4096; idx += 256) {
        int row = idx >> 8, h = idx & 255;
        Abuf[row * 512 + 256 + h] = (__bf16)stagef[idx];
    }
    __syncthreads();

    int wave = tid >> 5, lane = tid & 31;
    int n0 = wave * 32;
    v8f acc0 = {}, acc1 = {};
#pragma unroll 4
    for (int kb = 0; kb < 16; ++kb) {
        int k0 = kb * 32;
        v16bf a  = load_frag(Abuf + k0, 512, lane);
        v16bf b0 = load_frag(Wcat + n0 * 512 + k0, 512, lane);
        v16bf b1 = load_frag(Wcat + (n0 + 16) * 512 + k0, 512, lane);
        acc0 = __builtin_amdgcn_wmma_f32_16x16x32_bf16(
            false, a, false, b0, (short)0, acc0, false, false);
        acc1 = __builtin_amdgcn_wmma_f32_16x16x32_bf16(
            false, a, false, b1, (short)0, acc1, false, false);
    }
    int rbase = (lane >> 4) << 3;
#pragma unroll
    for (int sub = 0; sub < 2; ++sub) {
        int col  = n0 + sub * 16 + (lane & 15);
        float hb = hbase[b * 256 + col];
        const v8f& acc = sub ? acc1 : acc0;
#pragma unroll
        for (int rr = 0; rr < 8; ++rr)
            hid[(rbase + rr) * 256 + col] = fmaxf(acc[rr] + hb, 0.f);
    }
    __syncthreads();

    if (tid < 128) {
        int row = tid >> 3, e = tid & 7;
        float s = b2[e];
        for (int m = 0; m < 256; ++m) s = fmaf(hid[row * 256 + m], w2s[m * 8 + e], s);
        int j = j0 + row;
        float mk = (float)mask[(b * 64 + i) * 64 + j];
        out[(((b * 64 + i) * 64 + j) * 8) + e] = s * mk;
    }
}

// ---------------- host launcher ----------------
extern "C" void kernel_launch(void* const* d_in, const int* in_sizes, int n_in,
                              void* d_out, int out_size, void* d_ws, size_t ws_size,
                              hipStream_t stream) {
    const float* feat_body  = (const float*)d_in[0];
    const float* feat_img_1 = (const float*)d_in[1];
    const float* feat_img_2 = (const float*)d_in[2];
    const float* feat_ssl   = (const float*)d_in[3];
    const int*   full_mask  = (const int*)  d_in[4];
    const float* img_fc_w   = (const float*)d_in[5];
    const float* img_fc_b   = (const float*)d_in[6];
    const float* node_fc_w  = (const float*)d_in[7];
    const float* node_fc_b  = (const float*)d_in[8];
    const float* edge_fc_w1 = (const float*)d_in[9];
    const float* edge_fc_b1 = (const float*)d_in[10];
    const float* edge_fc_w2 = (const float*)d_in[11];
    const float* edge_fc_b2 = (const float*)d_in[12];

    char* ws = (char*)d_ws;
    float*  prop    = (float*)ws;  ws += (size_t)1024 * 256 * 4;
    float*  msgs    = (float*)ws;  ws += (size_t)5 * 1024 * 256 * 4;
    __bf16* Wimg_t  = (__bf16*)ws; ws += (size_t)256 * 256 * 2;
    __bf16* Wnode_t = (__bf16*)ws; ws += (size_t)5 * 256 * 256 * 2;
    __bf16* Wcat    = (__bf16*)ws; ws += (size_t)256 * 512 * 2;
    float*  fi2p    = (float*)ws;  ws += (size_t)16 * 256 * 4;
    float*  fsslp   = (float*)ws;  ws += (size_t)16 * 256 * 4;
    float*  hbase   = (float*)ws;  ws += (size_t)16 * 256 * 4;

    // --- setup: weight transposes / folds (all tiny) ---
    k_transpose_bf16<<<256, 256, 0, stream>>>(img_fc_w, Wimg_t, 256, 256);
    for (int t = 0; t < T_; ++t)
        k_transpose_bf16<<<256, 256, 0, stream>>>(node_fc_w + t * 65536,
                                                  Wnode_t + t * 65536, 256, 256);
    k_wcat_a<<<256, 256, 0, stream>>>(edge_fc_w1, Wcat);
    k_wcat_b<<<256, 256, 0, stream>>>(img_fc_w, edge_fc_w1, Wcat);
    k_small_proj<<<dim3(16, 2), 256, 0, stream>>>(feat_img_2, feat_ssl, img_fc_w,
                                                  img_fc_b, fi2p, fsslp);
    k_hbase<<<16, 256, 0, stream>>>(fi2p, fsslp, img_fc_b, edge_fc_w1, edge_fc_b1, hbase);

    // --- prop0 = feat_body @ W_img + b ---
    k_gemm_act<<<64, 256, 0, stream>>>(feat_body, Wimg_t, img_fc_b, prop);

    // --- T propagation steps ---
    for (int t = 0; t < T_; ++t) {
        k_gemm_act<<<64, 256, 0, stream>>>(prop, Wnode_t + t * 65536,
                                           node_fc_b + t * 256, msgs + t * 1024 * 256);
        k_agg<<<1024, 256, 0, stream>>>(msgs + t * 1024 * 256, full_mask, prop);
    }

    // --- fused edge scoring ---
    k_edge<<<4096, 256, 0, stream>>>(msgs, feat_img_1, Wcat, hbase,
                                     edge_fc_w2, edge_fc_b2, full_mask, (float*)d_out);
}
